// Encoder3_74998718923370
// MI455X (gfx1250) — matrix operand order repaired and checked by reading, not compiled, not run on hardware
//
#include <hip/hip_runtime.h>
#include <hip/hip_bf16.h>

// ---------------------------------------------------------------------------
// CDNA5 (gfx1250) GCN encoder.
//  - Dense GEMMs via v_wmma_f32_16x16x32_bf16, one wave32 per 16x16 C tile.
//  - Graph gather/scatter via f32 global atomics (L2-resident: all activation
//    arrays are 25.6MB each, total working set ~165MB < 192MB L2).
// ---------------------------------------------------------------------------

typedef __bf16 bf16_t;
typedef __attribute__((ext_vector_type(16))) __bf16        v16bf;
typedef __attribute__((ext_vector_type(8)))  float         v8f;
typedef __attribute__((ext_vector_type(4)))  unsigned int  u32x4;

union Frag16 { v16bf v; u32x4 q[2]; };

__device__ __forceinline__ bf16_t f32_to_bf16(float f) {
    unsigned u = __builtin_bit_cast(unsigned, f);
    u += 0x7FFFu + ((u >> 16) & 1u);            // round-to-nearest-even
    unsigned short h = (unsigned short)(u >> 16);
    return __builtin_bit_cast(bf16_t, h);
}

// ---------------------------------------------------------------------------
// Elementwise / graph kernels
// ---------------------------------------------------------------------------

__global__ void k_zero_f32(float* __restrict__ p, long n) {
    long i = (long)blockIdx.x * blockDim.x + threadIdx.x;
    if (i < n) p[i] = 0.0f;
}

__global__ void k_degree(const int* __restrict__ dst, float* __restrict__ deg, int E) {
    int e = blockIdx.x * blockDim.x + threadIdx.x;
    if (e < E) atomicAdd(&deg[dst[e]], 1.0f);
}

// deg -> rsqrt(deg + 1)  (in place)
__global__ void k_dis(float* __restrict__ deg, int N) {
    int i = blockIdx.x * blockDim.x + threadIdx.x;
    if (i < N) deg[i] = rsqrtf(deg[i] + 1.0f);
}

__global__ void k_cvt_bf16(const float* __restrict__ s, bf16_t* __restrict__ d, long n) {
    long i = (long)blockIdx.x * blockDim.x + threadIdx.x;
    if (i < n) d[i] = f32_to_bf16(s[i]);
}

// W [K x Nc] f32 row-major  ->  Wt [Nc x K] bf16 row-major (transposed)
__global__ void k_wt(const float* __restrict__ W, bf16_t* __restrict__ Wt, int K, int Nc) {
    int idx = blockIdx.x * blockDim.x + threadIdx.x;
    if (idx < K * Nc) {
        int k = idx / Nc;
        int n = idx - k * Nc;
        Wt[(size_t)n * K + k] = f32_to_bf16(W[idx]);
    }
}

// one 128-thread block per edge: agg[dst] += hW[src] * dis[src]*dis[dst]
__global__ void k_scatter(const int* __restrict__ src, const int* __restrict__ dst,
                          const float* __restrict__ dis, const float* __restrict__ hW,
                          float* __restrict__ agg, int H) {
    int e = blockIdx.x;
    int f = threadIdx.x;
    int s = src[e], d = dst[e];
    float w = dis[s] * dis[d];
    atomicAdd(&agg[(size_t)d * H + f], hW[(size_t)s * H + f] * w);
}

// out = relu(hin + agg + hW * dis^2 + b)   (residual + self-loop + bias)
__global__ void k_layer_finish(const float* __restrict__ hin, const float* __restrict__ agg,
                               const float* __restrict__ hW, const float* __restrict__ dis,
                               const float* __restrict__ b, float* __restrict__ out, int H) {
    int i = blockIdx.x;
    int f = threadIdx.x;
    size_t idx = (size_t)i * H + f;
    float ds = dis[i];
    float v = hin[idx] + agg[idx] + hW[idx] * ds * ds + b[f];
    out[idx] = fmaxf(v, 0.0f);
}

// one 128-thread block per node: accumulate JK-concat features into pooled sums
__global__ void k_pool(const float* __restrict__ r1, const float* __restrict__ r2,
                       const float* __restrict__ r3, const int* __restrict__ batch,
                       float* __restrict__ pooled, float* __restrict__ cnt, int H) {
    int i = blockIdx.x;
    int f = threadIdx.x;
    int b = batch[i];
    size_t nb = (size_t)i * H + f;
    size_t pb = (size_t)b * (3 * H);
    atomicAdd(&pooled[pb + f],         r1[nb]);
    atomicAdd(&pooled[pb + H + f],     r2[nb]);
    atomicAdd(&pooled[pb + 2 * H + f], r3[nb]);
    if (f == 0) atomicAdd(&cnt[b], 1.0f);
}

__global__ void k_pool_div(float* __restrict__ pooled, const float* __restrict__ cnt,
                           int G, int H3) {
    int idx = blockIdx.x * blockDim.x + threadIdx.x;
    if (idx < G * H3) pooled[idx] /= fmaxf(cnt[idx / H3], 1.0f);
}

// one 128-thread block per graph row: L2 normalize
__global__ void k_normalize(const float* __restrict__ p, float* __restrict__ out, int P) {
    __shared__ float s[128];
    int g = blockIdx.x;
    int f = threadIdx.x;
    float v = p[(size_t)g * P + f];
    s[f] = v * v;
    __syncthreads();
    for (int st = blockDim.x >> 1; st > 0; st >>= 1) {
        if (f < st) s[f] += s[f + st];
        __syncthreads();
    }
    out[(size_t)g * P + f] = v / fmaxf(sqrtf(s[0]), 1e-12f);
}

// ---------------------------------------------------------------------------
// WMMA GEMM: C[M x Nc] = A[M x K](bf16) * Bt[Nc x K](bf16, pre-transposed)
//            + bias (optional) , optional relu.  One wave32 per 16x16 tile.
// K consumed in 32-wide steps with v_wmma_f32_16x16x32_bf16.
// ---------------------------------------------------------------------------
__global__ void k_gemm_bf16_wmma(const bf16_t* __restrict__ A,
                                 const bf16_t* __restrict__ Bt,
                                 const float* __restrict__ bias,
                                 float* __restrict__ C,
                                 int M, int K, int Nc, int relu) {
    const int lane = threadIdx.x;          // 0..31 (wave32)
    const int rowBase = blockIdx.x * 16;
    const int colBase = blockIdx.y * 16;
    const int l15 = lane & 15;
    const int hi  = (lane >> 4) & 1;       // lane half: 0 -> lanes 0-15, 1 -> 16-31

    const int aRow = rowBase + l15;        // A fragment: row M = lane%16
    const int bCol = colBase + l15;        // B fragment: col N = lane%16
    const int ga = hi ? 8 : 0;             // A K-group base (ISA 16-bit A 16x32 layout)
    const int gb = hi ? 8 : 0;             // B K-group base (same ISA layout as A)

    const bf16_t* aPtr = A  + (size_t)aRow * K;
    const bf16_t* bPtr = Bt + (size_t)bCol * K;

    v8f c = {};
    for (int k0 = 0; k0 < K; k0 += 32) {
        Frag16 fa, fb;
        fa.q[0] = *(const u32x4*)(aPtr + k0 + ga);        // K = g..g+7
        fa.q[1] = *(const u32x4*)(aPtr + k0 + 16 + ga);   // K = 16+g..16+g+7
        fb.q[0] = *(const u32x4*)(bPtr + k0 + gb);        // K = gb..gb+7
        fb.q[1] = *(const u32x4*)(bPtr + k0 + 16 + gb);   // K = 16+gb..16+gb+7
        c = __builtin_amdgcn_wmma_f32_16x16x32_bf16(
                /*neg_a=*/false, fa.v, /*neg_b=*/false, fb.v,
                /*c_mod=*/(short)0, c, /*reuse_a=*/false, /*reuse_b=*/false);
    }

    const float bv = bias ? bias[bCol] : 0.0f;
    const int mBase = rowBase + (hi ? 8 : 0);            // C layout: VGPR r -> M=r(+8)
    for (int r = 0; r < 8; ++r) {
        float v = c[r] + bv;
        if (relu) v = fmaxf(v, 0.0f);
        C[(size_t)(mBase + r) * Nc + bCol] = v;
    }
}

// ---------------------------------------------------------------------------
// Host launcher
// ---------------------------------------------------------------------------
extern "C" void kernel_launch(void* const* d_in, const int* in_sizes, int n_in,
                              void* d_out, int out_size, void* d_ws, size_t ws_size,
                              hipStream_t stream) {
    const float* x     = (const float*)d_in[0];
    const int*   ei    = (const int*)  d_in[1];
    const int*   batch = (const int*)  d_in[2];
    const float* W_in  = (const float*)d_in[3];
    const float* b_in  = (const float*)d_in[4];
    const float* W1    = (const float*)d_in[5];
    const float* b1    = (const float*)d_in[6];
    const float* W2    = (const float*)d_in[7];
    const float* b2    = (const float*)d_in[8];
    const float* W3    = (const float*)d_in[9];
    const float* b3    = (const float*)d_in[10];
    const float* Wp1   = (const float*)d_in[11];
    const float* bp1   = (const float*)d_in[12];
    const float* Wp2   = (const float*)d_in[13];
    const float* bp2   = (const float*)d_in[14];
    float* out = (float*)d_out;

    const int E  = in_sizes[1] / 2;
    const int N  = in_sizes[2];
    const int H  = in_sizes[4];          // 128
    const int F  = in_sizes[0] / N;      // 128
    const int P  = in_sizes[14];         // 128
    const int G  = out_size / P;         // 512
    const int H3 = 3 * H;                // 384

    const int* src = ei;
    const int* dst = ei + E;

    // ---- workspace carve-up (256B aligned) ----
    char* ws = (char*)d_ws;
    size_t off = 0;
    auto alloc = [&](size_t bytes) -> char* {
        char* p = ws + off;
        off = (off + bytes + 255) & ~(size_t)255;
        return p;
    };
    float*  dis       = (float*) alloc((size_t)N * 4);
    float*  h         = (float*) alloc((size_t)N * H * 4);
    float*  hW        = (float*) alloc((size_t)N * H * 4);
    float*  agg       = (float*) alloc((size_t)N * H * 4);
    float*  r1        = (float*) alloc((size_t)N * H * 4);
    float*  r2        = (float*) alloc((size_t)N * H * 4);
    float*  r3        = (float*) alloc((size_t)N * H * 4);
    bf16_t* act_bf    = (bf16_t*)alloc((size_t)N * H * 2);
    bf16_t* Wint      = (bf16_t*)alloc((size_t)F * H * 2);
    bf16_t* W1t       = (bf16_t*)alloc((size_t)H * H * 2);
    bf16_t* W2t       = (bf16_t*)alloc((size_t)H * H * 2);
    bf16_t* W3t       = (bf16_t*)alloc((size_t)H * H * 2);
    bf16_t* Wp1t      = (bf16_t*)alloc((size_t)H3 * H * 2);
    bf16_t* Wp2t      = (bf16_t*)alloc((size_t)H * P * 2);
    float*  pooled    = (float*) alloc((size_t)G * H3 * 4);
    float*  cnt       = (float*) alloc((size_t)G * 4);
    bf16_t* pooled_bf = (bf16_t*)alloc((size_t)G * H3 * 2);
    float*  ph        = (float*) alloc((size_t)G * H * 4);
    bf16_t* ph_bf     = (bf16_t*)alloc((size_t)G * H * 2);
    float*  p         = (float*) alloc((size_t)G * P * 4);
    (void)ws_size; (void)n_in;

    const int TB = 256;
    auto blocks = [&](long n) { return (unsigned)((n + TB - 1) / TB); };

    // ---- degrees & symmetric norm ----
    k_zero_f32<<<blocks(N), TB, 0, stream>>>(dis, N);
    k_degree  <<<blocks(E), TB, 0, stream>>>(dst, dis, E);
    k_dis     <<<blocks(N), TB, 0, stream>>>(dis, N);

    // ---- weight prep: transpose + bf16 (tiny) ----
    k_wt<<<blocks((long)F  * H), TB, 0, stream>>>(W_in, Wint, F,  H);
    k_wt<<<blocks((long)H  * H), TB, 0, stream>>>(W1,   W1t,  H,  H);
    k_wt<<<blocks((long)H  * H), TB, 0, stream>>>(W2,   W2t,  H,  H);
    k_wt<<<blocks((long)H  * H), TB, 0, stream>>>(W3,   W3t,  H,  H);
    k_wt<<<blocks((long)H3 * H), TB, 0, stream>>>(Wp1,  Wp1t, H3, H);
    k_wt<<<blocks((long)H  * P), TB, 0, stream>>>(Wp2,  Wp2t, H,  P);

    // ---- input projection: h = x @ W_in + b_in ----
    k_cvt_bf16<<<blocks((long)N * F), TB, 0, stream>>>(x, act_bf, (long)N * F);
    k_gemm_bf16_wmma<<<dim3(N / 16, H / 16), 32, 0, stream>>>(
        act_bf, Wint, b_in, h, N, F, H, /*relu=*/0);

    // ---- 3 residual GCN layers ----
    auto run_layer = [&](const float* hin, const bf16_t* Wt, const float* bvec, float* hout) {
        k_cvt_bf16<<<blocks((long)N * H), TB, 0, stream>>>(hin, act_bf, (long)N * H);
        k_gemm_bf16_wmma<<<dim3(N / 16, H / 16), 32, 0, stream>>>(
            act_bf, Wt, (const float*)nullptr, hW, N, H, H, /*relu=*/0);
        k_zero_f32<<<blocks((long)N * H), TB, 0, stream>>>(agg, (long)N * H);
        k_scatter<<<E, H, 0, stream>>>(src, dst, dis, hW, agg, H);
        k_layer_finish<<<N, H, 0, stream>>>(hin, agg, hW, dis, bvec, hout, H);
    };
    run_layer(h,  W1t, b1, r1);
    run_layer(r1, W2t, b2, r2);
    run_layer(r2, W3t, b3, r3);

    // ---- JK-concat + global mean pool ----
    k_zero_f32<<<blocks((long)G * H3), TB, 0, stream>>>(pooled, (long)G * H3);
    k_zero_f32<<<blocks(G), TB, 0, stream>>>(cnt, G);
    k_pool    <<<N, H, 0, stream>>>(r1, r2, r3, batch, pooled, cnt, H);
    k_pool_div<<<blocks((long)G * H3), TB, 0, stream>>>(pooled, cnt, G, H3);

    // ---- projection head + L2 normalize ----
    k_cvt_bf16<<<blocks((long)G * H3), TB, 0, stream>>>(pooled, pooled_bf, (long)G * H3);
    k_gemm_bf16_wmma<<<dim3(G / 16, H / 16), 32, 0, stream>>>(
        pooled_bf, Wp1t, bp1, ph, G, H3, H, /*relu=*/1);
    k_cvt_bf16<<<blocks((long)G * H), TB, 0, stream>>>(ph, ph_bf, (long)G * H);
    k_gemm_bf16_wmma<<<dim3(G / 16, P / 16), 32, 0, stream>>>(
        ph_bf, Wp2t, bp2, p, G, H, P, /*relu=*/0);
    k_normalize<<<G, P, 0, stream>>>(p, out, P);
}
